// GNN_65824668778682
// MI455X (gfx1250) — matrix operand order repaired and checked by reading, not compiled, not run on hardware
//
#include <hip/hip_runtime.h>
#include <hip/hip_bf16.h>

#define NN 20000
#define NE 640000
#define NCH ((NN + 255) / 256)   // 79 scan chunks

typedef __attribute__((ext_vector_type(16))) __bf16 v16bf;
typedef __attribute__((ext_vector_type(8)))  __bf16 v8bf;
typedef __attribute__((ext_vector_type(4)))  __bf16 v4bf;
typedef __attribute__((ext_vector_type(8)))  float  v8f;

static __device__ __forceinline__ __bf16 f2bf(float f) {
    // round-to-nearest-even fp32 -> bf16
    unsigned u = __builtin_bit_cast(unsigned, f);
    u += 0x7FFFu + ((u >> 16) & 1u);
    unsigned short s = (unsigned short)(u >> 16);
    return __builtin_bit_cast(__bf16, s);
}

// ---------------- norm pipeline ----------------

__global__ void k_zero(float* __restrict__ p, int n) {
    int i = blockIdx.x * 256 + threadIdx.x;
    if (i < n) p[i] = 0.0f;
}

__global__ void k_zeroi(int* __restrict__ p, int n) {
    int i = blockIdx.x * 256 + threadIdx.x;
    if (i < n) p[i] = 0;
}

__global__ void k_deg(const long long* __restrict__ ei, const float* __restrict__ ew,
                      float* __restrict__ deg) {
    int e = blockIdx.x * 256 + threadIdx.x;
    if (e < NE) atomicAdd(&deg[(int)ei[NE + e]], ew[e]);
}

__global__ void k_dinv(float* __restrict__ deg) {
    int n = blockIdx.x * 256 + threadIdx.x;
    if (n < NN) {
        float d = deg[n];
        deg[n] = (d > 0.0f) ? rsqrtf(d) : 0.0f;
    }
}

__global__ void k_norm(const long long* __restrict__ ei, const float* __restrict__ ew,
                       const float* __restrict__ dinv, float* __restrict__ norm) {
    int e = blockIdx.x * 256 + threadIdx.x;
    if (e < NE) norm[e] = dinv[(int)ei[e]] * ew[e] * dinv[(int)ei[NE + e]];
}

// ---------------- device-side CSR-by-dst build ----------------

__global__ void k_count(const long long* __restrict__ ei, int* __restrict__ cnt) {
    int e = blockIdx.x * 256 + threadIdx.x;
    if (e < NE) atomicAdd(&cnt[(int)ei[NE + e]], 1);
}

// per-chunk sums (LDS tree reduction)
__global__ void k_scan_sum(const int* __restrict__ cnt, int* __restrict__ P) {
    __shared__ int sm[256];
    int i = blockIdx.x * 256 + threadIdx.x;
    sm[threadIdx.x] = (i < NN) ? cnt[i] : 0;
    __syncthreads();
#pragma unroll
    for (int s = 128; s > 0; s >>= 1) {
        if (threadIdx.x < s) sm[threadIdx.x] += sm[threadIdx.x + s];
        __syncthreads();
    }
    if (threadIdx.x == 0) P[blockIdx.x] = sm[0];
}

// exclusive scan of the 79 chunk totals (tiny, single thread)
__global__ void k_scan_tops(int* __restrict__ P) {
    if (blockIdx.x == 0 && threadIdx.x == 0) {
        int s = 0;
        for (int b = 0; b < NCH; ++b) { int v = P[b]; P[b] = s; s += v; }
    }
}

// per-chunk exclusive scan (LDS Hillis-Steele) + chunk offset -> rowptr
__global__ void k_scan_chunks(const int* __restrict__ cnt, const int* __restrict__ P,
                              int* __restrict__ rowptr) {
    __shared__ int sm[256];
    int i = blockIdx.x * 256 + threadIdx.x;
    int v = (i < NN) ? cnt[i] : 0;
    sm[threadIdx.x] = v;
    __syncthreads();
#pragma unroll
    for (int s = 1; s < 256; s <<= 1) {
        int add = (threadIdx.x >= s) ? sm[threadIdx.x - s] : 0;
        __syncthreads();
        sm[threadIdx.x] += add;
        __syncthreads();
    }
    int incl = sm[threadIdx.x];
    if (i < NN) rowptr[i] = P[blockIdx.x] + incl - v;   // exclusive
    if (i == NN - 1) rowptr[NN] = P[blockIdx.x] + incl; // total = NE
}

__global__ void k_copyi(const int* __restrict__ a, int* __restrict__ b, int n) {
    int i = blockIdx.x * 256 + threadIdx.x;
    if (i < n) b[i] = a[i];
}

__global__ void k_fill(const long long* __restrict__ ei, const float* __restrict__ norm,
                       int* __restrict__ wp, int* __restrict__ esrc,
                       float* __restrict__ enorm) {
    int e = blockIdx.x * 256 + threadIdx.x;
    if (e < NE) {
        int s = (int)ei[e];
        int d = (int)ei[NE + e];
        int pos = atomicAdd(&wp[d], 1);
        esrc[pos]  = s;
        enorm[pos] = norm[e];
    }
}

// ---------------- propagation (gather-style SpMM hops, no atomics) ----------------
// Hcat layout: [NN, 4*FIN] fp32 row-major; hop k occupies columns [k*FIN, (k+1)*FIN)

template <int FIN>
__global__ void k_copy_block0(const float* __restrict__ in, float* __restrict__ Hcat) {
    int i = blockIdx.x * 256 + threadIdx.x;
    if (i < NN * FIN) {
        int n = i / FIN, f = i % FIN;
        Hcat[(size_t)n * (4 * FIN) + f] = in[i];
    }
}

template <int FIN>
__global__ void k_hop_csr(const int* __restrict__ rowptr, const int* __restrict__ esrc,
                          const float* __restrict__ enorm, float* __restrict__ Hcat,
                          int kblk) {
    constexpr int CH = FIN / 4;          // float4 chunks per node row
    int t = blockIdx.x * 256 + threadIdx.x;
    if (t >= NN * CH) return;
    int d = t / CH;
    int j = (t % CH) * 4;
    const int beg = rowptr[d], end = rowptr[d + 1];
    float4 acc = make_float4(0.f, 0.f, 0.f, 0.f);
    const float* __restrict__ src_base = Hcat + (size_t)(kblk - 1) * FIN + j;
    for (int i = beg; i < end; ++i) {
        const int   s = esrc[i];
        const float w = enorm[i];
        const float4 v = *(const float4*)(src_base + (size_t)s * (4 * FIN));
        acc.x += w * v.x; acc.y += w * v.y; acc.z += w * v.z; acc.w += w * v.w;
    }
    *(float4*)&Hcat[(size_t)d * (4 * FIN) + kblk * FIN + j] = acc;
}

// ---------------- operand conversion passes ----------------

template <int KTOT>
__global__ void k_h2bf(const float* __restrict__ H, __bf16* __restrict__ Hb) {
    int i = blockIdx.x * 256 + threadIdx.x;        // one float4 per thread
    if (i < NN * KTOT / 4) {
        const float4 v = ((const float4*)H)[i];
        v4bf o;
        o[0] = f2bf(v.x); o[1] = f2bf(v.y); o[2] = f2bf(v.z); o[3] = f2bf(v.w);
        *(v4bf*)(Hb + 4 * (size_t)i) = o;
    }
}

// W fp32 [KTOT, FOUT] row-major -> Wt bf16 [FOUT, KTOT] row-major
template <int KTOT, int FOUT>
__global__ void k_wt(const float* __restrict__ W, __bf16* __restrict__ Wt) {
    int i = blockIdx.x * 256 + threadIdx.x;
    if (i < KTOT * FOUT) {
        int k = i / FOUT, n = i % FOUT;
        Wt[(size_t)n * KTOT + k] = f2bf(W[i]);
    }
}

// ---------------- fused 4-hop GEMM: out = Hb @ Wt^T + b (WMMA bf16) ----------------

template <int NT>
struct Frag {
    v8bf  alo, ahi;
    v16bf b[NT];
};

template <int NT>
static __device__ __forceinline__ void frag_load(Frag<NT>& f,
                                                 const __bf16* __restrict__ Arow,
                                                 const __bf16* __restrict__ const* Brow,
                                                 int k0) {
    f.alo = *(const v8bf*)(Arow + k0);
    f.ahi = *(const v8bf*)(Arow + k0 + 8);
#pragma unroll
    for (int t = 0; t < NT; ++t) f.b[t] = *(const v16bf*)(Brow[t] + k0);
}

template <int NT>
static __device__ __forceinline__ void frag_mma(const Frag<NT>& f, v8f* acc) {
    const v16bf a = __builtin_shufflevector(f.alo, f.ahi,
        0, 1, 2, 3, 4, 5, 6, 7, 8, 9, 10, 11, 12, 13, 14, 15);
#pragma unroll
    for (int t = 0; t < NT; ++t)
        acc[t] = __builtin_amdgcn_wmma_f32_16x16x32_bf16(
            false, a, false, f.b[t], (short)0, acc[t], false, false);
}

template <int FIN, int FOUT, int NT, bool LRELU>
__global__ __launch_bounds__(128) void k_gemm(const __bf16* __restrict__ A,   // [NN, 4*FIN]
                                              const __bf16* __restrict__ Wt,  // [FOUT, 4*FIN]
                                              const float* __restrict__ bias,
                                              float* __restrict__ out) {      // [NN, FOUT]
    constexpr int KTOT   = 4 * FIN;   // multiple of 64, >= 128
    constexpr int CHUNKS = FOUT / (16 * NT);
    const int wave = threadIdx.x >> 5;
    const int lane = threadIdx.x & 31;
    const int gid  = blockIdx.x * 4 + wave;
    const int total = (NN / 16) * CHUNKS;
    if (gid >= total) return;                 // wave-uniform: EXEC stays all-ones
    const int mtile  = gid / CHUNKS;
    const int nchunk = gid % CHUNKS;
    const int m0 = mtile * 16;
    const int n0 = nchunk * (NT * 16);
    const int half = lane >> 4;               // 0: lanes 0-15, 1: lanes 16-31
    const int lidx = lane & 15;

    v8f acc[NT];
#pragma unroll
    for (int t = 0; t < NT; ++t) acc[t] = (v8f)(0.0f);

    const __bf16* __restrict__ Arow = A + (size_t)(m0 + lidx) * KTOT + 16 * half;
    const __bf16* __restrict__ Brow[NT];
#pragma unroll
    for (int t = 0; t < NT; ++t)
        Brow[t] = Wt + (size_t)(n0 + t * 16 + lidx) * KTOT + 16 * half;

    Frag<NT> f0, f1;
    frag_load<NT>(f0, Arow, Brow, 0);
    frag_load<NT>(f1, Arow, Brow, 32);

    for (int k0 = 0; k0 < KTOT - 64; k0 += 64) {
        __builtin_prefetch(Arow + k0 + 128, 0, 3);   // global_prefetch_b8
        frag_mma<NT>(f0, acc);
        frag_load<NT>(f0, Arow, Brow, k0 + 64);
        frag_mma<NT>(f1, acc);
        frag_load<NT>(f1, Arow, Brow, k0 + 96);
    }
    frag_mma<NT>(f0, acc);
    frag_mma<NT>(f1, acc);

#pragma unroll
    for (int t = 0; t < NT; ++t) {
        const int nn = n0 + t * 16 + lidx;
        const float bv = bias[nn];
#pragma unroll
        for (int r = 0; r < 8; ++r) {
            const int mm = m0 + r + half * 8;
            float v = acc[t][r] + bv;
            if (LRELU) v = (v > 0.0f) ? v : 0.01f * v;
            out[(size_t)mm * FOUT + nn] = v;
        }
    }
}

// ---------------- host side ----------------

static inline int nb256(long long n) { return (int)((n + 255) / 256); }

template <int FIN, int FOUT, int NT, bool LR>
static void run_layer(const float* hin, const float* W, const float* b, float* hout,
                      const int* rowptr, const int* esrc, const float* enorm,
                      float* Hcat, __bf16* Hb, __bf16* Wt, hipStream_t stream) {
    constexpr int KTOT = 4 * FIN;
    k_copy_block0<FIN><<<nb256((long long)NN * FIN), 256, 0, stream>>>(hin, Hcat);
    for (int k = 1; k <= 3; ++k)
        k_hop_csr<FIN><<<nb256((long long)NN * (FIN / 4)), 256, 0, stream>>>(
            rowptr, esrc, enorm, Hcat, k);
    k_h2bf<KTOT><<<nb256((long long)NN * KTOT / 4), 256, 0, stream>>>(Hcat, Hb);
    k_wt<KTOT, FOUT><<<nb256((long long)KTOT * FOUT), 256, 0, stream>>>(W, Wt);
    constexpr int CHUNKS = FOUT / (16 * NT);
    const int waves = (NN / 16) * CHUNKS;
    k_gemm<FIN, FOUT, NT, LR><<<(waves + 3) / 4, 128, 0, stream>>>(Hb, Wt, b, hout);
}

extern "C" void kernel_launch(void* const* d_in, const int* in_sizes, int n_in,
                              void* d_out, int out_size, void* d_ws, size_t ws_size,
                              hipStream_t stream) {
    const float*     x  = (const float*)d_in[0];
    const long long* ei = (const long long*)d_in[1];   // int64 edge_index [2, NE]
    const float*     ew = (const float*)d_in[2];
    const float* Wp[4]  = {(const float*)d_in[3], (const float*)d_in[5],
                           (const float*)d_in[7], (const float*)d_in[9]};
    const float* bp[4]  = {(const float*)d_in[4], (const float*)d_in[6],
                           (const float*)d_in[8], (const float*)d_in[10]};
    float* out = (float*)d_out;

    // workspace layout (4-byte units from base)
    float* ws     = (float*)d_ws;
    float* norm   = ws;                         // 640,000 f
    float* deg    = ws + 640000;                // 20,000 f (deg -> dinv in place)
    int*   cnt    = (int*)(ws + 660000);        // 20,000 i
    int*   wp     = (int*)(ws + 680000);        // 20,000 i
    int*   P      = (int*)(ws + 700000);        // 128 i (chunk totals)
    int*   rowptr = (int*)(ws + 700200);        // 20,001 i
    int*   esrc   = (int*)(ws + 720256);        // 640,000 i
    float* enorm  = ws + 1360256;               // 640,000 f
    float* Hcat   = ws + 2000256;               // 20,000*1024 f (16B aligned)
    float* outA   = Hcat + (size_t)NN * 1024;   // 20,000*256 f
    float* outB   = outA + (size_t)NN * 256;    // 20,000*256 f
    __bf16* Hb    = (__bf16*)(outB + (size_t)NN * 256); // 20,000*1024 bf16 (32B aligned)
    __bf16* Wt    = Hb + (size_t)NN * 1024;             // up to 256*1024 bf16

    // gcn_norm
    k_zero<<<nb256(NN), 256, 0, stream>>>(deg, NN);
    k_deg<<<nb256(NE), 256, 0, stream>>>(ei, ew, deg);
    k_dinv<<<nb256(NN), 256, 0, stream>>>(deg);
    k_norm<<<nb256(NE), 256, 0, stream>>>(ei, ew, deg, norm);

    // CSR-by-dst build: histogram -> hierarchical exclusive scan -> fill
    k_zeroi<<<nb256(NN), 256, 0, stream>>>(cnt, NN);
    k_count<<<nb256(NE), 256, 0, stream>>>(ei, cnt);
    k_scan_sum<<<NCH, 256, 0, stream>>>(cnt, P);
    k_scan_tops<<<1, 32, 0, stream>>>(P);
    k_scan_chunks<<<NCH, 256, 0, stream>>>(cnt, P, rowptr);
    k_copyi<<<nb256(NN), 256, 0, stream>>>(rowptr, wp, NN);
    k_fill<<<nb256(NE), 256, 0, stream>>>(ei, norm, wp, esrc, enorm);

    // 4 TAGConv layers; leaky_relu(0.01) after layers 0 and 1 only
    run_layer<64, 256, 4, true >(x,    Wp[0], bp[0], outA, rowptr, esrc, enorm, Hcat, Hb, Wt, stream);
    run_layer<256, 256, 4, true >(outA, Wp[1], bp[1], outB, rowptr, esrc, enorm, Hcat, Hb, Wt, stream);
    run_layer<256, 256, 4, false>(outB, Wp[2], bp[2], outA, rowptr, esrc, enorm, Hcat, Hb, Wt, stream);
    run_layer<256, 32, 2, false>(outA, Wp[3], bp[3], out,  rowptr, esrc, enorm, Hcat, Hb, Wt, stream);
}
